// attentionLayer_37855841747246
// MI455X (gfx1250) — matrix operand order repaired and checked by reading, not compiled, not run on hardware
//
#include <hip/hip_runtime.h>

// ---------------------------------------------------------------------------
// Flash attention (causal) for MI455X / gfx1250, wave32, WMMA bf16 path.
// B=16, Sq=Sk=2048, D=64, fp32 in/out.  D = A*B + C via v_wmma_f32_16x16x32_bf16.
// Compute-bound (~17 GFLOP vs ~34 MB of HBM traffic), so both GEMMs run on the
// WMMA pipe in bf16 with f32 accumulation; softmax stays fp32 on the VALU.
// ---------------------------------------------------------------------------

typedef __attribute__((ext_vector_type(16))) __bf16 v16bf;
typedef __attribute__((ext_vector_type(2)))  __bf16 v2bf;
typedef __attribute__((ext_vector_type(8)))  float  v8f;
typedef __attribute__((ext_vector_type(4)))  float  f4;

#define BATCH   16
#define SQ      2048
#define SK      2048
#define DH      64
#define BM      128          // query rows per workgroup (8 waves x 16 rows)
#define BN      64           // key/value columns per tile
#define MASKN   (-4294967295.0f)   // reference _masking_num

// Native bf16 conversion (gfx1250 has first-class bf16; lowers to HW cvt ops)
__device__ __forceinline__ __bf16 to_bf16(float x) { return (__bf16)x; }

// WMMA 16-bit fragment swizzle (ISA 7.12.2, 16-bit A 16x32 table; B mirrored):
// element (row16, kk) with kk in [0,32) lives at lane = row16 + 16*((kk>>3)&1),
// vector element e = (kk&7) + 8*((kk>>4)&1).  Returns bf16-element offset
// inside a 512-element (1 KB) fragment.
__device__ __forceinline__ unsigned swz(unsigned r16, unsigned kk) {
    return (r16 + 16u * ((kk >> 3) & 1u)) * 16u + (kk & 7u) + 8u * ((kk >> 4) & 1u);
}

__global__ __launch_bounds__(256) void
attn_fwd_wmma(const float* __restrict__ Q, const float* __restrict__ K,
              const float* __restrict__ V, float* __restrict__ O) {
    // LDS: K B-frags [0,4096), V B-frags [4096,8192), per-wave P A-frags [8192,16384)
    __shared__ __align__(128) unsigned short lds[16384];   // 32 KB (of 320 KB/WGP)

    const int tid  = threadIdx.x;
    const int lane = tid & 31;
    const int wave = tid >> 5;
    const int mrow = lane & 15;     // N (or M) index within 16-lane half
    const int half = lane >> 4;     // which 16-lane half of the wave
    const int q0   = blockIdx.x * BM;
    const int b    = blockIdx.y;

    // ---- Q A-fragments: 16x32 each, loaded straight from HBM in reg layout ----
    // inverse swizzle: kk = (e&7) + 8*half + 16*(e>>3)
    v16bf qf[2];
    {
        const float* Qb = Q + ((size_t)b * SQ + (size_t)(q0 + wave * 16 + mrow)) * DH;
        #pragma unroll
        for (int c = 0; c < 2; ++c)
            #pragma unroll
            for (int e = 0; e < 16; ++e) {
                int kk = (e & 7) + 8 * half + 16 * (e >> 3);
                qf[c][e] = to_bf16(Qb[c * 32 + kk]);
            }
    }

    // ---- flash-attention running state (per lane: 8 rows of the 16x16 C tile) ----
    v8f   o[4] = {};                   // 16 x 64 output accumulator (C layout)
    float mi[8], li[8];
    #pragma unroll
    for (int v = 0; v < 8; ++v) { mi[v] = -1.0e30f; li[v] = 0.0f; }

    const int ntiles = q0 / BN + 2;    // causal: only tiles touching the diagonal
    const int ldrow  = tid >> 2;       // cooperative loader: row within 64-row tile
    const int lddc   = (tid & 3) * 16; // 16-column slab per thread
    const int rbase  = q0 + wave * 16; // first query row owned by this wave

    for (int jt = 0; jt < ntiles; ++jt) {
        const int kb = jt * BN;

        // ---- cooperative HBM -> LDS load of K and V tiles, pre-swizzled ----
        #pragma unroll
        for (int mat = 0; mat < 2; ++mat) {
            const float* src = (mat ? V : K) +
                ((size_t)b * SK + (size_t)(kb + ldrow)) * DH + lddc;
            const unsigned base = mat ? 4096u : 0u;
            #pragma unroll
            for (int g = 0; g < 4; ++g) {
                f4 w = *(const f4*)(src + g * 4);
                float vv[4] = { w.x, w.y, w.z, w.w };
                #pragma unroll
                for (int i = 0; i < 4; i += 2) {
                    int d = lddc + g * 4 + i;                       // even -> pairable
                    v2bf pk; pk[0] = to_bf16(vv[i]); pk[1] = to_bf16(vv[i + 1]);
                    if (mat == 0) {
                        unsigned frag = (unsigned)((d >> 5) * 4 + (ldrow >> 4));
                        unsigned off  = base + frag * 512u + swz((unsigned)(ldrow & 15), (unsigned)(d & 31));
                        *(v2bf*)&lds[off] = pk;                     // ds_store_b32
                    } else {
                        #pragma unroll
                        for (int j = 0; j < 2; ++j) {
                            int dd = d + j;
                            unsigned frag = (unsigned)((ldrow >> 5) * 4 + ((dd >> 4) & 3));
                            unsigned off  = base + frag * 512u + swz((unsigned)(dd & 15), (unsigned)(ldrow & 31));
                            *(__bf16*)&lds[off] = pk[j];
                        }
                    }
                }
            }
            if (jt + 1 < ntiles)   // warm L2/WGP$ for the next tile
                __builtin_prefetch(src + (size_t)BN * DH, 0, 0);
        }
        __syncthreads();

        // ---- S = (Q K^T): 2 k-chunks x 4 n-blocks of v_wmma_f32_16x16x32_bf16 ----
        v8f s[4] = {};
        #pragma unroll
        for (int kc = 0; kc < 2; ++kc) {
            #pragma unroll
            for (int nb = 0; nb < 4; ++nb) {
                v16bf bfrag = *(const v16bf*)&lds[(unsigned)(kc * 4 + nb) * 512u + (unsigned)lane * 16u];
                s[nb] = __builtin_amdgcn_wmma_f32_16x16x32_bf16(
                            false, qf[kc], false, bfrag, (short)0, s[nb], false, false);
            }
        }

        // ---- scale + padding mask + causal mask, track row max ----
        // Causal compares only needed when this tile can cross the wave's
        // diagonal: wave-uniform condition, safe to branch around (no barriers).
        const bool needCausal = (kb + BN - 1) > rbase;
        float rm[8];
        #pragma unroll
        for (int v = 0; v < 8; ++v) {
            const int r = rbase + v + 8 * half;
            float best = -3.0e38f;
            #pragma unroll
            for (int nb = 0; nb < 4; ++nb) {
                float x = s[nb][v] * 0.125f;          // 1/sqrt(64)
                if (x == 0.0f) x += MASKN;            // reference "padding" mask
                if (needCausal) {
                    const int c = kb + nb * 16 + mrow;
                    if (c > r) x += MASKN;            // causal mask
                }
                s[nb][v] = x;
                best = fmaxf(best, x);
            }
            rm[v] = best;
        }
        // reduce across the 16 lanes holding this row (xor 1,2,4,8 stays in-half)
        #pragma unroll
        for (int d = 8; d >= 1; d >>= 1)
            #pragma unroll
            for (int v = 0; v < 8; ++v)
                rm[v] = fmaxf(rm[v], __shfl_xor(rm[v], d, 32));

        float mnew[8], alpha[8];
        #pragma unroll
        for (int v = 0; v < 8; ++v) {
            mnew[v]  = fmaxf(mi[v], rm[v]);
            alpha[v] = __expf(mi[v] - mnew[v]);
        }

        // ---- P = exp(S - m), write to per-wave LDS region in A-fragment swizzle ----
        const unsigned pbase = 8192u + (unsigned)wave * 1024u;
        float rs[8] = {};
        #pragma unroll
        for (int nb = 0; nb < 4; ++nb) {
            #pragma unroll
            for (int v = 0; v < 8; ++v) {
                float p = __expf(s[nb][v] - mnew[v]);
                rs[v] += p;
                const int col = nb * 16 + mrow;               // k index within BN
                const int m   = v + 8 * half;                 // row within 16
                unsigned off = pbase + (unsigned)(col >> 5) * 512u +
                               swz((unsigned)m, (unsigned)(col & 31));
                *(__bf16*)&lds[off] = to_bf16(p);             // ds_store_b16
            }
        }
        #pragma unroll
        for (int d = 8; d >= 1; d >>= 1)
            #pragma unroll
            for (int v = 0; v < 8; ++v)
                rs[v] += __shfl_xor(rs[v], d, 32);

        #pragma unroll
        for (int v = 0; v < 8; ++v) {
            li[v] = li[v] * alpha[v] + rs[v];
            mi[v] = mnew[v];
        }
        #pragma unroll
        for (int nb = 0; nb < 4; ++nb)
            #pragma unroll
            for (int v = 0; v < 8; ++v)
                o[nb][v] *= alpha[v];

        // ---- O += P * V : 2 k-chunks x 4 n-blocks of bf16 WMMA ----
        // (same-wave LDS RAW: LDS ops from one wave stay in order; compiler
        //  inserts the s_wait_dscnt for the fragment reads below)
        #pragma unroll
        for (int kc = 0; kc < 2; ++kc) {
            v16bf pfrag = *(const v16bf*)&lds[pbase + (unsigned)kc * 512u + (unsigned)lane * 16u];
            #pragma unroll
            for (int nb = 0; nb < 4; ++nb) {
                v16bf vfrag = *(const v16bf*)&lds[4096u + (unsigned)(kc * 4 + nb) * 512u + (unsigned)lane * 16u];
                o[nb] = __builtin_amdgcn_wmma_f32_16x16x32_bf16(
                            false, pfrag, false, vfrag, (short)0, o[nb], false, false);
            }
        }
        __syncthreads();   // K/V LDS tiles are overwritten next iteration
    }

    // ---- epilogue: O /= l, store fp32 (C layout: row = v + 8*half, col = nb*16+mrow)
    float* Ob = O + (size_t)b * SQ * DH;
    #pragma unroll
    for (int v = 0; v < 8; ++v) {
        const float inv = 1.0f / li[v];
        const int   r   = q0 + wave * 16 + v + 8 * half;
        #pragma unroll
        for (int nb = 0; nb < 4; ++nb)
            Ob[(size_t)r * DH + nb * 16 + mrow] = o[nb][v] * inv;
    }
}

extern "C" void kernel_launch(void* const* d_in, const int* in_sizes, int n_in,
                              void* d_out, int out_size, void* d_ws, size_t ws_size,
                              hipStream_t stream) {
    (void)in_sizes; (void)n_in; (void)d_ws; (void)ws_size; (void)out_size;
    const float* q = (const float*)d_in[0];
    const float* k = (const float*)d_in[1];
    const float* v = (const float*)d_in[2];
    // d_in[3] = masks (int32) -- unused by the reference math
    float* out = (float*)d_out;

    dim3 grid(SQ / BM, BATCH);   // 16 x 16 workgroups
    dim3 block(256);             // 8 wave32 waves
    attn_fwd_wmma<<<grid, block, 0, stream>>>(q, k, v, out);
}
